// LigerLMHeadCE_4887672783290
// MI455X (gfx1250) — matrix operand order repaired and checked by reading, not compiled, not run on hardware
//
#include <hip/hip_runtime.h>

// ---------------------------------------------------------------------------
// Fused LM-head cross-entropy for MI455X (gfx1250, wave32, WMMA bf16).
//   logits = input[BT,H] @ weight[V,H]^T   (fp32 -> bf16 inputs, fp32 accum)
//   loss   = mean over valid rows of (logsumexp(logits_row) - logits[row,tgt])
// WG tile 128x128, wave tile 32x64 (2x4 WMMA frags), K staged 64-deep through
// double-buffered LDS via GLOBAL_LOAD_ASYNC_TO_LDS_B128 (ASYNCcnt-tracked).
// ---------------------------------------------------------------------------

#define BT 4096
#define H  4096
#define V  32000
#define TILE_M 128
#define TILE_N 128
#define TILE_K 64
#define NCHUNK (V / TILE_N)      // 250
#define NK     (H / TILE_K)      // 64
#define LDA    72                // padded LDS row stride (halves): 64 + 8
#define ABUF   (TILE_M * LDA)
#define BBUF   (TILE_N * LDA)
#define IGNORE_INDEX (-100)

typedef __attribute__((ext_vector_type(8)))  float   v8f;
typedef __attribute__((ext_vector_type(8)))  __bf16  v8bf;
typedef __attribute__((ext_vector_type(16))) __bf16  v16bf;

#if __has_builtin(__builtin_amdgcn_global_load_async_to_lds_b128)
#define HAS_ASYNC_LDS 1
// Exact parameter type per hipcc diagnostic:
//   '__attribute__((__vector_size__(4 * sizeof(int)))) int *'
typedef int v4i_vs __attribute__((__vector_size__(4 * sizeof(int))));
#if __has_builtin(__builtin_amdgcn_s_wait_asynccnt)
#define WAIT_ASYNC(n) __builtin_amdgcn_s_wait_asynccnt(n)
#else
#define WAIT_ASYNC(n) asm volatile("s_wait_asynccnt %0" ::"i"(n) : "memory")
#endif
#else
#define HAS_ASYNC_LDS 0
#endif

// ---------------------------------------------------------------------------
// Kernel 0: fp32 -> bf16 (round-to-nearest-even) for input and weight.
// Also zeroes the loss/count accumulators (thread 0).
// ---------------------------------------------------------------------------
__global__ void ce_cvt_kernel(const float* __restrict__ inF,
                              const float* __restrict__ wF,
                              unsigned short* __restrict__ inB,
                              unsigned short* __restrict__ wB,
                              float* __restrict__ sums) {
    size_t idx = (size_t)blockIdx.x * blockDim.x + threadIdx.x;
    if (idx == 0) { sums[0] = 0.0f; sums[1] = 0.0f; }
    const size_t nIn = (size_t)BT * H;
    const size_t nW  = (size_t)V * H;
    const size_t stride = (size_t)gridDim.x * blockDim.x;
    for (size_t i = idx; i < nIn + nW; i += stride) {
        float f = (i < nIn) ? inF[i] : wF[i - nIn];
        unsigned u = __float_as_uint(f);
        u += 0x7FFFu + ((u >> 16) & 1u);   // RNE to bf16
        unsigned short h = (unsigned short)(u >> 16);
        if (i < nIn) inB[i] = h; else wB[i - nIn] = h;
    }
}

// ---------------------------------------------------------------------------
// Kernel 1: tiled bf16 WMMA GEMM + fused partial log-softmax.
// grid = (V/TILE_N, BT/TILE_M), block = 256 threads = 8 wave32s.
// waveM in [0,4): 32 M-rows each; waveN in [0,2): 64 N-cols each.
// ---------------------------------------------------------------------------
__global__ __launch_bounds__(256)
void ce_gemm_lse_kernel(const unsigned short* __restrict__ inB,
                        const unsigned short* __restrict__ wB,
                        const long long* __restrict__ target,
                        float* __restrict__ m_part,
                        float* __restrict__ s_part,
                        float* __restrict__ tArr) {
    __shared__ __align__(32) unsigned short As[2 * ABUF];  // 36 KB
    __shared__ __align__(32) unsigned short Bs[2 * BBUF];  // 36 KB
    __shared__ float2 stats[TILE_M][2];                    //  2 KB

    const int tid    = threadIdx.x;
    const int lane   = tid & 31;
    const int l16    = lane & 15;
    const int hl     = lane >> 4;          // half-lane group (0 or 1)
    const int waveId = tid >> 5;
    const int waveM  = waveId & 3;         // 4 slots of 32 rows
    const int waveN  = waveId >> 2;        // 2 slots of 64 cols

    const int m_base = blockIdx.y * TILE_M;
    const int n_base = blockIdx.x * TILE_N;

    // Cooperative staging: 128 rows x 64 halves = 16 KB per matrix per stage.
    // 2 threads/row, 64 B (4 x b128) per thread for each of A and B.
    const int sRow  = tid >> 1;            // 0..127
    const int sHalf = tid & 1;             // K-half (32 halves = 64 B)
    const unsigned short* aSrcBase =
        inB + (size_t)(m_base + sRow) * H + sHalf * 32;
    const unsigned short* bSrcBase =
        wB  + (size_t)(n_base + sRow) * H + sHalf * 32;
    const int aDst = sRow * LDA + sHalf * 32;
    const int bDst = sRow * LDA + sHalf * 32;

    auto stage = [&](int kt, int buf) {
        const int k0 = kt * TILE_K;
#if HAS_ASYNC_LDS
        v4i_vs* aS = (v4i_vs*)(aSrcBase + k0);
        v4i_vs* bS = (v4i_vs*)(bSrcBase + k0);
        v4i_vs* aD = (v4i_vs*)&As[buf * ABUF + aDst];
        v4i_vs* bD = (v4i_vs*)&Bs[buf * BBUF + bDst];
        __builtin_amdgcn_global_load_async_to_lds_b128(aS, aD,  0, 0);
        __builtin_amdgcn_global_load_async_to_lds_b128(aS, aD, 16, 0);
        __builtin_amdgcn_global_load_async_to_lds_b128(aS, aD, 32, 0);
        __builtin_amdgcn_global_load_async_to_lds_b128(aS, aD, 48, 0);
        __builtin_amdgcn_global_load_async_to_lds_b128(bS, bD,  0, 0);
        __builtin_amdgcn_global_load_async_to_lds_b128(bS, bD, 16, 0);
        __builtin_amdgcn_global_load_async_to_lds_b128(bS, bD, 32, 0);
        __builtin_amdgcn_global_load_async_to_lds_b128(bS, bD, 48, 0);
#else
        const uint4* aS = (const uint4*)(aSrcBase + k0);
        const uint4* bS = (const uint4*)(bSrcBase + k0);
        uint4* aD = (uint4*)&As[buf * ABUF + aDst];
        uint4* bD = (uint4*)&Bs[buf * BBUF + bDst];
        uint4 a0 = aS[0], a1 = aS[1], a2 = aS[2], a3 = aS[3];
        uint4 b0 = bS[0], b1 = bS[1], b2 = bS[2], b3 = bS[3];
        aD[0] = a0; aD[1] = a1; aD[2] = a2; aD[3] = a3;
        bD[0] = b0; bD[1] = b1; bD[2] = b2; bD[3] = b3;
#endif
    };

    v8f acc[2][4];
    #pragma unroll
    for (int ri = 0; ri < 2; ++ri)
        #pragma unroll
        for (int ci = 0; ci < 4; ++ci)
            acc[ri][ci] = (v8f)0.0f;

    stage(0, 0);
    for (int kt = 0; kt < NK; ++kt) {
        const int cur = kt & 1;
        if (kt + 1 < NK) {
            stage(kt + 1, cur ^ 1);
#if HAS_ASYNC_LDS
            WAIT_ASYNC(8);   // my 8 loads for `cur` retired (in-order)
#endif
        } else {
#if HAS_ASYNC_LDS
            WAIT_ASYNC(0);
#endif
        }
        __syncthreads();     // everyone's `cur` slab visible in LDS

        // Two 32-deep WMMA K-steps over the 64-deep slab.
        #pragma unroll
        for (int ks = 0; ks < 2; ++ks) {
            // A (16x32 bf16): lane l16 = row; lanes 0-15 hold K {0..7,16..23},
            //                 lanes 16-31 hold K {8..15,24..31}.
            v16bf aF[2];
            #pragma unroll
            for (int ri = 0; ri < 2; ++ri) {
                int row = waveM * 32 + ri * 16 + l16;
                int base = cur * ABUF + row * LDA + ks * 32 + 8 * hl;
                v8bf lo = *(const v8bf*)&As[base];
                v8bf hi = *(const v8bf*)&As[base + 16];
                aF[ri] = __builtin_shufflevector(lo, hi,
                    0, 1, 2, 3, 4, 5, 6, 7, 8, 9, 10, 11, 12, 13, 14, 15);
            }
            // B (32x16 bf16): lane l16 = col; lanes 0-15 hold K {0..7,16..23},
            //                 lanes 16-31 hold K {8..15,24..31}.
            v16bf bF[4];
            #pragma unroll
            for (int ci = 0; ci < 4; ++ci) {
                int col = waveN * 64 + ci * 16 + l16;
                int base = cur * BBUF + col * LDA + ks * 32 + 8 * hl;
                v8bf lo = *(const v8bf*)&Bs[base];
                v8bf hi = *(const v8bf*)&Bs[base + 16];
                bF[ci] = __builtin_shufflevector(lo, hi,
                    0, 1, 2, 3, 4, 5, 6, 7, 8, 9, 10, 11, 12, 13, 14, 15);
            }
            #pragma unroll
            for (int ri = 0; ri < 2; ++ri)
                #pragma unroll
                for (int ci = 0; ci < 4; ++ci)
                    acc[ri][ci] = __builtin_amdgcn_wmma_f32_16x16x32_bf16(
                        false, aF[ri], false, bF[ci],
                        (short)0, acc[ri][ci], false, false);
        }
        __syncthreads();     // reads of `cur` done before it is re-staged
    }

    // ------------------- fused partial log-softmax epilogue -----------------
    // C layout (16x16 f32): VGPR g, lane l -> row = g + 8*(l/16), col = l%16.
    #pragma unroll
    for (int ri = 0; ri < 2; ++ri) {
        #pragma unroll
        for (int g = 0; g < 8; ++g) {
            // local max over this wave's 4 column-fragments, then 16 lanes
            float lm = acc[ri][0][g];
            #pragma unroll
            for (int ci = 1; ci < 4; ++ci) lm = fmaxf(lm, acc[ri][ci][g]);
            float M = lm;
            #pragma unroll
            for (int off = 1; off < 16; off <<= 1)
                M = fmaxf(M, __shfl_xor(M, off, 32));
            float e = 0.0f;
            #pragma unroll
            for (int ci = 0; ci < 4; ++ci) e += __expf(acc[ri][ci][g] - M);
            float S = e;
            #pragma unroll
            for (int off = 1; off < 16; off <<= 1)
                S += __shfl_xor(S, off, 32);

            int rowLocal = waveM * 32 + ri * 16 + g + 8 * hl;
            if (l16 == 0) stats[rowLocal][waveN] = make_float2(M, S);

            // target-logit extraction: exactly one lane/WG owns (row, tgt)
            int rowG = m_base + rowLocal;
            long long tg = target[rowG];
            #pragma unroll
            for (int ci = 0; ci < 4; ++ci) {
                int colG = n_base + waveN * 64 + ci * 16 + l16;
                if (tg == (long long)colG) tArr[rowG] = acc[ri][ci][g];
            }
        }
    }
    __syncthreads();

    // Combine the two waveN partials per row; write chunk partials to HBM.
    if (tid < TILE_M) {
        float2 a = stats[tid][0];
        float2 b = stats[tid][1];
        float M = fmaxf(a.x, b.x);
        float S = a.y * __expf(a.x - M) + b.y * __expf(b.x - M);
        size_t o = (size_t)blockIdx.x * BT + (m_base + tid);
        m_part[o] = M;
        s_part[o] = S;
    }
}

// ---------------------------------------------------------------------------
// Kernel 2: per-row combine of the 250 chunk partials -> loss accumulation.
// ---------------------------------------------------------------------------
__global__ void ce_reduce_kernel(const float* __restrict__ m_part,
                                 const float* __restrict__ s_part,
                                 const float* __restrict__ tArr,
                                 const long long* __restrict__ target,
                                 float* __restrict__ sums) {
    int row = blockIdx.x * blockDim.x + threadIdx.x;
    if (row >= BT) return;
    float M = -INFINITY, S = 0.0f;
    for (int c = 0; c < NCHUNK; ++c) {
        float m = m_part[(size_t)c * BT + row];
        float s = s_part[(size_t)c * BT + row];
        float nM = fmaxf(M, m);
        S = S * __expf(M - nM) + s * __expf(m - nM);
        M = nM;
    }
    long long tg = target[row];
    if (tg != (long long)IGNORE_INDEX) {
        float loss = M + __logf(S) - tArr[row];
        atomicAdd(&sums[0], loss);
        atomicAdd(&sums[1], 1.0f);
    }
}

__global__ void ce_finalize_kernel(const float* __restrict__ sums,
                                   float* __restrict__ out) {
    if (blockIdx.x == 0 && threadIdx.x == 0)
        out[0] = sums[0] / fmaxf(sums[1], 1.0f);
}

// ---------------------------------------------------------------------------
// Launcher. Workspace layout (bytes):
//   [0)            bf16 input   BT*H*2      =  33,554,432
//   [+33554432)    bf16 weight  V*H*2       = 262,144,000
//   [+295698432)   m_part       NCHUNK*BT*4 =   4,096,000
//   [+299794432)   s_part       NCHUNK*BT*4 =   4,096,000
//   [+303890432)   tArr         BT*4        =      16,384
//   [+303906816)   sums         2*4
// ---------------------------------------------------------------------------
extern "C" void kernel_launch(void* const* d_in, const int* in_sizes, int n_in,
                              void* d_out, int out_size, void* d_ws, size_t ws_size,
                              hipStream_t stream) {
    const float*     inF    = (const float*)d_in[0];
    const float*     wF     = (const float*)d_in[1];
    const long long* target = (const long long*)d_in[2];
    float*           out    = (float*)d_out;

    char* ws = (char*)d_ws;
    unsigned short* inB    = (unsigned short*)(ws);
    unsigned short* wB     = (unsigned short*)(ws + 33554432ull);
    float*          m_part = (float*)(ws + 295698432ull);
    float*          s_part = (float*)(ws + 299794432ull);
    float*          tArr   = (float*)(ws + 303890432ull);
    float*          sums   = (float*)(ws + 303906816ull);

    // 0: convert to bf16 + zero accumulators
    ce_cvt_kernel<<<dim3(32768), dim3(256), 0, stream>>>(inF, wF, inB, wB, sums);

    // 1: fused GEMM + partial logsumexp
    dim3 grid(V / TILE_N, BT / TILE_M);   // (250, 32)
    ce_gemm_lse_kernel<<<grid, dim3(256), 0, stream>>>(inB, wB, target,
                                                       m_part, s_part, tArr);

    // 2: combine chunk partials per row
    ce_reduce_kernel<<<dim3((BT + 255) / 256), dim3(256), 0, stream>>>(
        m_part, s_part, tArr, target, sums);

    // 3: mean
    ce_finalize_kernel<<<dim3(1), dim3(1), 0, stream>>>(sums, out);
}
